// TextEmbedderLETS_89644557402360
// MI455X (gfx1250) — matrix operand order repaired and checked by reading, not compiled, not run on hardware
//
#include <hip/hip_runtime.h>

// ---------------- model dims ----------------
constexpr int Ln = 2, Dm = 2048, Hh = 16, DHn = 128, DFFn = 5504, BLKn = 64;
constexpr int Bn = 4, Sn = 1000, Cn = 7, Tn = 45;
constexpr int Mn = Bn * Sn;

// ---------------- types ----------------
typedef __attribute__((ext_vector_type(16))) __bf16 bf16x16;
typedef __attribute__((ext_vector_type(8)))  float  v8f;
typedef __attribute__((ext_vector_type(4)))  int    i4vec;

struct alignas(16) U4 { unsigned int x, y, z, w; };
union Frag { U4 u4[2]; bf16x16 v; };

#define AS1 __attribute__((address_space(1)))
#define AS3 __attribute__((address_space(3)))

#if defined(__AMDGCN__) && __has_builtin(__builtin_amdgcn_global_load_async_to_lds_b128)
#define HAVE_ASYNC_LDS 1
#else
#define HAVE_ASYNC_LDS 0
#endif

__device__ inline U4 u4zero() { U4 z; z.x = z.y = z.z = z.w = 0u; return z; }
__device__ inline v8f v8zero() {
  v8f z;
#pragma unroll
  for (int i = 0; i < 8; i++) z[i] = 0.f;
  return z;
}

__constant__ float NF4c[16] = {
  -1.0f, -0.6961928009986877f, -0.5250730514526367f, -0.39491748809814453f,
  -0.28444138169288635f, -0.18477343022823334f, -0.09105003625154495f, 0.0f,
  0.07958029955625534f, 0.16093020141124725f, 0.24611230194568634f,
  0.3379152417182922f, 0.44070982933044434f, 0.5626170039176941f,
  0.7229568362236023f, 1.0f };

// ---------------- embed gather: h[m,:] = embed[tok[m],:] ----------------
__global__ __launch_bounds__(256) void embed_kernel(const int* __restrict__ tok,
                                                    const float* __restrict__ emb,
                                                    float* __restrict__ h) {
  int m = blockIdx.x;
  int t = tok[m];
  const U4* src = (const U4*)(emb + (size_t)t * Dm);
  U4* dst = (U4*)(h + (size_t)m * Dm);
  for (int i = threadIdx.x; i < Dm / 4; i += 256) dst[i] = src[i];
}

// ---------------- rmsnorm: optional bf16 out, optional rstd out ----------------
__global__ __launch_bounds__(256) void rmsnorm_kernel(const float* __restrict__ h,
                                                      const float* __restrict__ w,
                                                      __bf16* __restrict__ out,
                                                      float* __restrict__ rstd_out) {
  int row = blockIdx.x;
  const float* hr = h + (size_t)row * Dm;
  float ss = 0.f;
  for (int d = threadIdx.x; d < Dm; d += 256) { float v = hr[d]; ss += v * v; }
  __shared__ float red[256];
  red[threadIdx.x] = ss;
  __syncthreads();
  for (int s = 128; s > 0; s >>= 1) {
    if (threadIdx.x < s) red[threadIdx.x] += red[threadIdx.x + s];
    __syncthreads();
  }
  float rstd = rsqrtf(red[0] * (1.f / Dm) + 1e-5f);
  if (out) {
    for (int d = threadIdx.x; d < Dm; d += 256)
      out[(size_t)row * Dm + d] = (__bf16)(hr[d] * rstd * w[d]);
  }
  if (rstd_out && threadIdx.x == 0) rstd_out[row] = rstd;
}

// ---------------- NF4 dequant -> bf16 [O, I] ----------------
__global__ __launch_bounds__(256) void dequant_kernel(const int* __restrict__ qidx,
                                                      const float* __restrict__ scale,
                                                      __bf16* __restrict__ out,
                                                      int O, int I) {
  int i = blockIdx.x * 256 + threadIdx.x;
  int o = blockIdx.y;
  if (i < I) {
    float v = NF4c[qidx[(size_t)o * I + i] & 15] * scale[(size_t)o * (I / BLKn) + (i >> 6)];
    out[(size_t)o * I + i] = (__bf16)v;
  }
}

// ---------------- GEMM: C[M,N] = A[M,K] * W[N,K]^T  (bf16 in, WMMA bf16) ----------
// Double-buffered LDS staging; async global->LDS path when the toolchain has it.
// MODE 0: Cf = v ; MODE 1: Cf += v (residual) ; MODE 2: Cb = (bf16)v
template <int MODE>
__global__ __launch_bounds__(256) void gemm_bf16_kernel(const __bf16* __restrict__ A,
                                                        const __bf16* __restrict__ W,
                                                        float* __restrict__ Cf,
                                                        __bf16* __restrict__ Cb,
                                                        int M, int N, int K) {
  __shared__ __bf16 As[2][128][40];  // +8 pad: 80B row stride, 16B aligned
  __shared__ __bf16 Ws[2][128][40];
  const int tid = threadIdx.x;
  const int wave = tid >> 5, lane = tid & 31, l16 = lane & 15, lh = lane >> 4;
  const int wm = wave >> 1, wn = wave & 1;
  const int bm0 = blockIdx.y * 128, bn0 = blockIdx.x * 128;

  // staging geometry: each thread moves 16 bf16 of A and 16 bf16 of W per tile
  const int lr = tid >> 1, lc = (tid & 1) * 16;
  const int arow = min(bm0 + lr, M - 1);  // clamp (guarded epilogue discards junk)
  const int wrow = min(bn0 + lr, N - 1);
  const __bf16* Abase = A + (size_t)arow * K + lc;
  const __bf16* Wbase = W + (size_t)wrow * K + lc;

  v8f acc[2][4];
#pragma unroll
  for (int i = 0; i < 2; i++)
#pragma unroll
    for (int j = 0; j < 4; j++) acc[i][j] = v8zero();

  auto compute = [&](int buf) {
    Frag af[2], bf_[4];
#pragma unroll
    for (int mi = 0; mi < 2; mi++) {
      int r = wm * 32 + mi * 16 + l16;
      af[mi].u4[0] = *(const U4*)&As[buf][r][lh * 8];
      af[mi].u4[1] = *(const U4*)&As[buf][r][lh * 8 + 16];
    }
#pragma unroll
    for (int ni = 0; ni < 4; ni++) {
      int r = wn * 64 + ni * 16 + l16;
      bf_[ni].u4[0] = *(const U4*)&Ws[buf][r][lh * 8];
      bf_[ni].u4[1] = *(const U4*)&Ws[buf][r][lh * 8 + 16];
    }
#pragma unroll
    for (int mi = 0; mi < 2; mi++)
#pragma unroll
      for (int ni = 0; ni < 4; ni++)
        acc[mi][ni] = __builtin_amdgcn_wmma_f32_16x16x32_bf16(
            false, af[mi].v, false, bf_[ni].v, (short)0, acc[mi][ni], false, false);
  };

  const int nsteps = K / 32;

#if HAVE_ASYNC_LDS
  auto stage_async = [&](int buf, int k0) {
    __builtin_amdgcn_global_load_async_to_lds_b128(
        (AS1 i4vec*)(Abase + k0), (AS3 i4vec*)&As[buf][lr][lc], 0, 0);
    __builtin_amdgcn_global_load_async_to_lds_b128(
        (AS1 i4vec*)(Abase + k0 + 8), (AS3 i4vec*)&As[buf][lr][lc + 8], 0, 0);
    __builtin_amdgcn_global_load_async_to_lds_b128(
        (AS1 i4vec*)(Wbase + k0), (AS3 i4vec*)&Ws[buf][lr][lc], 0, 0);
    __builtin_amdgcn_global_load_async_to_lds_b128(
        (AS1 i4vec*)(Wbase + k0 + 8), (AS3 i4vec*)&Ws[buf][lr][lc + 8], 0, 0);
  };
  stage_async(0, 0);
  for (int it = 0; it < nsteps; ++it) {
    const int cur = it & 1;
    const bool more = (it + 1 < nsteps);
    if (more) {
      stage_async(cur ^ 1, (it + 1) * 32);
      asm volatile("s_wait_asynccnt 0x4" ::: "memory");  // current tile retired
    } else {
      asm volatile("s_wait_asynccnt 0x0" ::: "memory");
    }
    __syncthreads();
    compute(cur);
    __syncthreads();  // all waves done reading before next async overwrite
  }
#else
  U4 aR0, aR1, wR0, wR1;
  auto gload = [&](int k0) {
    const __bf16* pa = Abase + k0;
    aR0 = *(const U4*)pa;
    aR1 = *(const U4*)(pa + 8);
    const __bf16* pw = Wbase + k0;
    wR0 = *(const U4*)pw;
    wR1 = *(const U4*)(pw + 8);
  };
  auto lstore = [&](int buf) {
    *(U4*)&As[buf][lr][lc] = aR0;
    *(U4*)&As[buf][lr][lc + 8] = aR1;
    *(U4*)&Ws[buf][lr][lc] = wR0;
    *(U4*)&Ws[buf][lr][lc + 8] = wR1;
  };
  gload(0);
  lstore(0);
  for (int it = 0; it < nsteps; ++it) {
    const int cur = it & 1;
    const bool more = (it + 1 < nsteps);
    if (more) gload((it + 1) * 32);  // overlap global latency with WMMA below
    __syncthreads();
    compute(cur);
    if (more) lstore(cur ^ 1);
  }
#endif

#pragma unroll
  for (int mi = 0; mi < 2; mi++)
#pragma unroll
    for (int ni = 0; ni < 4; ni++)
#pragma unroll
      for (int r = 0; r < 8; r++) {
        int gr = bm0 + wm * 32 + mi * 16 + r + 8 * lh;
        int gc = bn0 + wn * 64 + ni * 16 + l16;
        if (gr < M && gc < N) {
          float v = acc[mi][ni][r];
          size_t o = (size_t)gr * N + gc;
          if (MODE == 0) Cf[o] = v;
          else if (MODE == 1) Cf[o] += v;
          else Cb[o] = (__bf16)v;
        }
      }
}

// ---------------- rotary (in-place on bf16 q,k) ----------------
__global__ __launch_bounds__(256) void rotary_kernel(__bf16* __restrict__ qb,
                                                     __bf16* __restrict__ kb) {
  int i = blockIdx.x * 256 + threadIdx.x;
  if (i >= Mn * Hh * 64) return;
  int j = i & 63;
  int hh = (i >> 6) & (Hh - 1);
  int m = i >> 10;
  int s = m % Sn;
  float inv = __powf(10000.f, -(float)(2 * j) * (1.f / (float)DHn));
  float ang = (float)s * inv;
  float sn, cs;
  __sincosf(ang, &sn, &cs);
  size_t base = (size_t)m * Dm + hh * DHn + j;
  float q0 = (float)qb[base], q1 = (float)qb[base + 64];
  qb[base] = (__bf16)(q0 * cs - q1 * sn);
  qb[base + 64] = (__bf16)(q1 * cs + q0 * sn);
  float k0 = (float)kb[base], k1 = (float)kb[base + 64];
  kb[base] = (__bf16)(k0 * cs - k1 * sn);
  kb[base + 64] = (__bf16)(k1 * cs + k0 * sn);
}

// ---------------- flash attention (causal), WMMA bf16 ----------------
__global__ __launch_bounds__(128) void flash_attn_kernel(const __bf16* __restrict__ Q,
                                                         const __bf16* __restrict__ Kg,
                                                         const __bf16* __restrict__ Vg,
                                                         __bf16* __restrict__ ctx) {
  __shared__ __bf16 Kt[32][DHn + 8];
  __shared__ __bf16 Vt[32][DHn + 8];
  __shared__ __bf16 Ps[4][16][40];
  const int tid = threadIdx.x, wave = tid >> 5, lane = tid & 31;
  const int l16 = lane & 15, lh = lane >> 4;
  const int bh = blockIdx.y, b = bh / Hh, hh = bh % Hh;
  const int qr0 = blockIdx.x * 64 + wave * 16;

  // Q fragments (4 slices of 32 along DH=128), A-matrix layout
  Frag qf[4];
  {
    int qs = qr0 + l16;
    if (qs < Sn) {
      const __bf16* qp = Q + ((size_t)(b * Sn + qs)) * Dm + hh * DHn;
#pragma unroll
      for (int ks = 0; ks < 4; ks++) {
        qf[ks].u4[0] = *(const U4*)(qp + ks * 32 + lh * 8);
        qf[ks].u4[1] = *(const U4*)(qp + ks * 32 + lh * 8 + 16);
      }
    } else {
#pragma unroll
      for (int ks = 0; ks < 4; ks++) { qf[ks].u4[0] = u4zero(); qf[ks].u4[1] = u4zero(); }
    }
  }

  v8f o[8];
#pragma unroll
  for (int t = 0; t < 8; t++) o[t] = v8zero();
  float mrow[8], lrow[8];
#pragma unroll
  for (int r = 0; r < 8; r++) { mrow[r] = -__builtin_inff(); lrow[r] = 0.f; }

  const float scl = 0.08838834764831845f;  // 1/sqrt(128)
  const int kend = min(Sn, (int)blockIdx.x * 64 + 64);

  for (int kt0 = 0; kt0 < kend; kt0 += 32) {
    __syncthreads();
    // stage 32x128 K and V tiles
#pragma unroll
    for (int j = 0; j < 4; j++) {
      int idx = tid * 4 + j;
      int r = idx >> 4;
      int c = (idx & 15) * 8;
      int krow = kt0 + r;
      U4 kv = u4zero(), vv = u4zero();
      if (krow < Sn) {
        kv = *(const U4*)(Kg + ((size_t)(b * Sn + krow)) * Dm + hh * DHn + c);
        vv = *(const U4*)(Vg + ((size_t)(b * Sn + krow)) * Dm + hh * DHn + c);
      }
      *(U4*)&Kt[r][c] = kv;
      *(U4*)&Vt[r][c] = vv;
      // hint the next tile into cache while we compute this one
      int nrow = krow + 32;
      if (nrow < Sn) {
        __builtin_prefetch(Kg + ((size_t)(b * Sn + nrow)) * Dm + hh * DHn + c, 0, 0);
        __builtin_prefetch(Vg + ((size_t)(b * Sn + nrow)) * Dm + hh * DHn + c, 0, 0);
      }
    }
    __syncthreads();

    // scores: two 16x16 tiles (32 keys)
    v8f sacc[2];
    sacc[0] = v8zero(); sacc[1] = v8zero();
#pragma unroll
    for (int kt = 0; kt < 2; kt++) {
#pragma unroll
      for (int ks = 0; ks < 4; ks++) {
        Frag kf;
        int kr = kt * 16 + l16;
        kf.u4[0] = *(const U4*)&Kt[kr][ks * 32 + lh * 8];
        kf.u4[1] = *(const U4*)&Kt[kr][ks * 32 + lh * 8 + 16];
        sacc[kt] = __builtin_amdgcn_wmma_f32_16x16x32_bf16(
            false, qf[ks].v, false, kf.v, (short)0, sacc[kt], false, false);
      }
    }
    // scale + causal mask
#pragma unroll
    for (int kt = 0; kt < 2; kt++)
#pragma unroll
      for (int r = 0; r < 8; r++) {
        int qrow = qr0 + r + 8 * lh;
        int kpos = kt0 + kt * 16 + l16;
        float v = sacc[kt][r] * scl;
        if (kpos > qrow || kpos >= Sn || qrow >= Sn) v = -1e30f;
        sacc[kt][r] = v;
      }
    // online softmax (row reductions across 16 lanes)
#pragma unroll
    for (int r = 0; r < 8; r++) {
      float mx = fmaxf(sacc[0][r], sacc[1][r]);
#pragma unroll
      for (int m = 1; m < 16; m <<= 1) mx = fmaxf(mx, __shfl_xor(mx, m, 32));
      float mnew = fmaxf(mrow[r], mx);
      float alpha = __expf(mrow[r] - mnew);
      mrow[r] = mnew;
      float p0 = __expf(sacc[0][r] - mnew);
      float p1 = __expf(sacc[1][r] - mnew);
      sacc[0][r] = p0; sacc[1][r] = p1;
      float ps = p0 + p1;
#pragma unroll
      for (int m = 1; m < 16; m <<= 1) ps += __shfl_xor(ps, m, 32);
      lrow[r] = lrow[r] * alpha + ps;
#pragma unroll
      for (int t = 0; t < 8; t++) o[t][r] *= alpha;
    }
    // P: C-layout -> LDS -> A-layout
#pragma unroll
    for (int kt = 0; kt < 2; kt++)
#pragma unroll
      for (int r = 0; r < 8; r++)
        Ps[wave][r + 8 * lh][kt * 16 + l16] = (__bf16)sacc[kt][r];
    asm volatile("s_wait_dscnt 0x0" ::: "memory");
    Frag pf;
    pf.u4[0] = *(const U4*)&Ps[wave][l16][lh * 16];
    pf.u4[1] = *(const U4*)&Ps[wave][l16][lh * 16 + 8];
    // O += P(16x32) x V(32x128)
#pragma unroll
    for (int t = 0; t < 8; t++) {
      Frag vf;
      int dh = t * 16 + l16;
#pragma unroll
      for (int j = 0; j < 8; j++) {
        vf.v[2 * j] = Vt[lh * 16 + 2 * j][dh];
        vf.v[2 * j + 1] = Vt[lh * 16 + 2 * j + 1][dh];
      }
      o[t] = __builtin_amdgcn_wmma_f32_16x16x32_bf16(
          false, pf.v, false, vf.v, (short)0, o[t], false, false);
    }
  }

#pragma unroll
  for (int r = 0; r < 8; r++) {
    int qrow = qr0 + r + 8 * lh;
    if (qrow < Sn) {
      float inv = lrow[r] > 0.f ? 1.f / lrow[r] : 0.f;
#pragma unroll
      for (int t = 0; t < 8; t++)
        ctx[((size_t)(b * Sn + qrow)) * Dm + hh * DHn + t * 16 + l16] =
            (__bf16)(o[t][r] * inv);
    }
  }
}

// ---------------- silu(g)*u ----------------
__global__ __launch_bounds__(256) void silu_mul_kernel(const __bf16* __restrict__ g,
                                                       const __bf16* __restrict__ u,
                                                       __bf16* __restrict__ o, int n) {
  int i = blockIdx.x * 256 + threadIdx.x;
  if (i < n) {
    float gv = (float)g[i], uv = (float)u[i];
    o[i] = (__bf16)(gv * uv / (1.f + __expf(-gv)));
  }
}

// ---------------- final: mean over S of rmsnorm(h) ----------------
__global__ __launch_bounds__(256) void final_pool_kernel(const float* __restrict__ h,
                                                         const float* __restrict__ rstd,
                                                         const float* __restrict__ lnf,
                                                         float* __restrict__ out) {
  int i = blockIdx.x * 256 + threadIdx.x;
  if (i >= Bn * Dm) return;
  int b = i / Dm, d = i % Dm;
  float acc = 0.f;
  for (int s = 0; s < Sn; s++)
    acc += h[((size_t)(b * Sn + s)) * Dm + d] * rstd[b * Sn + s];
  out[i] = acc * lnf[d] * (1.f / (float)Sn);
}

// ---------------- series branch ----------------
__global__ __launch_bounds__(256) void series_mean_kernel(const float* __restrict__ x,
                                                          float* __restrict__ sm) {
  int i = blockIdx.x * 256 + threadIdx.x;
  if (i < Bn * Tn) {
    int b = i / Tn, t = i % Tn;
    float s = 0.f;
    for (int c = 0; c < Cn; c++) s += x[(size_t)(b * Cn + c) * Tn + t];
    sm[i] = s * (1.f / (float)Cn);
  }
}

__global__ __launch_bounds__(256) void series_add_kernel(const float* __restrict__ sm,
                                                         const float* __restrict__ pw,
                                                         const float* __restrict__ pb,
                                                         float* __restrict__ out) {
  int b = blockIdx.x, tid = threadIdx.x;
  float vals[8];
  float ss = 0.f;
#pragma unroll
  for (int j = 0; j < 8; j++) {
    int d = tid + j * 256;
    float a = pb[d];
    for (int t = 0; t < Tn; t++) a += sm[b * Tn + t] * pw[(size_t)d * Tn + t];
    vals[j] = a;
    ss += a * a;
  }
  __shared__ float red[256];
  red[tid] = ss;
  __syncthreads();
  for (int s2 = 128; s2 > 0; s2 >>= 1) {
    if (tid < s2) red[tid] += red[tid + s2];
    __syncthreads();
  }
  float inv = 1.f / fmaxf(sqrtf(red[0]), 1e-12f);
#pragma unroll
  for (int j = 0; j < 8; j++) {
    int d = tid + j * 256;
    out[(size_t)b * Dm + d] += vals[j] * inv;
  }
}

// ---------------- host orchestration ----------------
extern "C" void kernel_launch(void* const* d_in, const int* in_sizes, int n_in,
                              void* d_out, int out_size, void* d_ws, size_t ws_size,
                              hipStream_t stream) {
  (void)in_sizes; (void)n_in; (void)out_size; (void)ws_size;
  const float* x = (const float*)d_in[0];
  const int* tok = (const int*)d_in[1];
  const float* embed = (const float*)d_in[2];
  const int* widx[7] = { (const int*)d_in[3], (const int*)d_in[5], (const int*)d_in[7],
                         (const int*)d_in[9], (const int*)d_in[11], (const int*)d_in[13],
                         (const int*)d_in[15] };
  const float* wscl[7] = { (const float*)d_in[4], (const float*)d_in[6], (const float*)d_in[8],
                           (const float*)d_in[10], (const float*)d_in[12], (const float*)d_in[14],
                           (const float*)d_in[16] };
  const float* ln1 = (const float*)d_in[17];
  const float* ln2 = (const float*)d_in[18];
  const float* lnf = (const float*)d_in[19];
  const float* pw = (const float*)d_in[20];
  const float* pb = (const float*)d_in[21];
  float* out = (float*)d_out;

  char* wsp = (char*)d_ws;
  size_t off = 0;
  auto alloc = [&](size_t bytes) -> char* {
    char* p = wsp + off;
    off += (bytes + 255) & ~(size_t)255;
    return p;
  };
  float* h = (float*)alloc((size_t)Mn * Dm * 4);
  __bf16* xn = (__bf16*)alloc((size_t)Mn * Dm * 2);
  __bf16* qb = (__bf16*)alloc((size_t)Mn * Dm * 2);
  __bf16* kb = (__bf16*)alloc((size_t)Mn * Dm * 2);
  __bf16* vb = (__bf16*)alloc((size_t)Mn * Dm * 2);
  __bf16* ctx = (__bf16*)alloc((size_t)Mn * Dm * 2);
  __bf16* gb = (__bf16*)alloc((size_t)Mn * DFFn * 2);
  __bf16* ub = (__bf16*)alloc((size_t)Mn * DFFn * 2);
  __bf16* act = (__bf16*)alloc((size_t)Mn * DFFn * 2);
  __bf16* wdq = (__bf16*)alloc((size_t)DFFn * Dm * 2);
  float* rstd = (float*)alloc((size_t)Mn * 4);
  float* sm = (float*)alloc((size_t)Bn * Tn * 4);

  auto deq = [&](int w, int l, int O, int I) {
    dequant_kernel<<<dim3((I + 255) / 256, O), 256, 0, stream>>>(
        widx[w] + (size_t)l * O * I, wscl[w] + (size_t)l * O * (I / BLKn), wdq, O, I);
  };

  embed_kernel<<<Mn, 256, 0, stream>>>(tok, embed, h);

  dim3 gemmBlk(256);
  dim3 gDD((Dm + 127) / 128, (Mn + 127) / 128);
  dim3 gDF((DFFn + 127) / 128, (Mn + 127) / 128);

  for (int l = 0; l < Ln; l++) {
    rmsnorm_kernel<<<Mn, 256, 0, stream>>>(h, ln1 + (size_t)l * Dm, xn, nullptr);
    deq(0, l, Dm, Dm);
    gemm_bf16_kernel<2><<<gDD, gemmBlk, 0, stream>>>(xn, wdq, nullptr, qb, Mn, Dm, Dm);
    deq(1, l, Dm, Dm);
    gemm_bf16_kernel<2><<<gDD, gemmBlk, 0, stream>>>(xn, wdq, nullptr, kb, Mn, Dm, Dm);
    deq(2, l, Dm, Dm);
    gemm_bf16_kernel<2><<<gDD, gemmBlk, 0, stream>>>(xn, wdq, nullptr, vb, Mn, Dm, Dm);

    rotary_kernel<<<(Mn * Hh * 64 + 255) / 256, 256, 0, stream>>>(qb, kb);
    flash_attn_kernel<<<dim3((Sn + 63) / 64, Bn * Hh), 128, 0, stream>>>(qb, kb, vb, ctx);

    deq(3, l, Dm, Dm);
    gemm_bf16_kernel<1><<<gDD, gemmBlk, 0, stream>>>(ctx, wdq, h, nullptr, Mn, Dm, Dm);

    rmsnorm_kernel<<<Mn, 256, 0, stream>>>(h, ln2 + (size_t)l * Dm, xn, nullptr);
    deq(4, l, DFFn, Dm);
    gemm_bf16_kernel<2><<<gDF, gemmBlk, 0, stream>>>(xn, wdq, nullptr, gb, Mn, DFFn, Dm);
    deq(5, l, DFFn, Dm);
    gemm_bf16_kernel<2><<<gDF, gemmBlk, 0, stream>>>(xn, wdq, nullptr, ub, Mn, DFFn, Dm);
    silu_mul_kernel<<<((Mn * DFFn) + 255) / 256, 256, 0, stream>>>(gb, ub, act, Mn * DFFn);
    deq(6, l, Dm, DFFn);
    gemm_bf16_kernel<1><<<gDD, gemmBlk, 0, stream>>>(act, wdq, h, nullptr, Mn, Dm, DFFn);
  }

  rmsnorm_kernel<<<Mn, 256, 0, stream>>>(h, lnf, nullptr, rstd);
  final_pool_kernel<<<(Bn * Dm + 255) / 256, 256, 0, stream>>>(h, rstd, lnf, out);
  series_mean_kernel<<<1, 256, 0, stream>>>(x, sm);
  series_add_kernel<<<Bn, 256, 0, stream>>>(sm, pw, pb, out);
}
